// Attention_85856396247857
// MI455X (gfx1250) — compile-verified, matrix operand order checked by reading, not run on hardware
//
#include <hip/hip_runtime.h>
#include <hip/hip_bf16.h>
#include <cstdint>
#include <cstddef>

#define DIM    2048
#define HEADS  16
#define DHEAD  128
#define SEQ    2048
#define BATCH  2
#define MROWS  (BATCH * SEQ)   // 4096

typedef __attribute__((ext_vector_type(16))) __bf16 v16bf;
typedef __attribute__((ext_vector_type(8)))  float  v8f;
typedef __attribute__((ext_vector_type(4)))  unsigned int u32x4;
typedef __attribute__((ext_vector_type(8)))  int          i32x8;
typedef __attribute__((ext_vector_type(4)))  int          i32x4;

#ifndef __has_builtin
#define __has_builtin(x) 0
#endif
#if defined(__gfx1250__) && __has_builtin(__builtin_amdgcn_tensor_load_to_lds) && \
    __has_builtin(__builtin_amdgcn_s_wait_tensorcnt)
#define USE_TDM 1
#else
#define USE_TDM 0
#endif

union FragU { v16bf v; uint4 q[2]; };

// Load a 16x32 bf16 WMMA fragment from a row-major buffer (rows = M for A,
// rows = N for B fed as B^T), per the CDNA5 16-bit A/B VGPR layout:
// lane 0-15: row = lane,   K = {0..7} in VGPR0-3, {16..23} in VGPR4-7
// lane16-31: row = lane-16, K = {8..15} / {24..31}
__device__ __forceinline__ v16bf load_frag(const __bf16* base, int row0, int ld,
                                           int k0, int lane) {
  int r  = row0 + (lane & 15);
  int kb = k0 + ((lane & 16) >> 1);       // +0 or +8
  const __bf16* p = base + (size_t)r * ld + kb;
  FragU f;
  f.q[0] = *(const uint4*)(p);            // K = kb .. kb+7
  f.q[1] = *(const uint4*)(p + 16);       // K = kb+16 .. kb+23
  return f.v;
}

__device__ __forceinline__ v8f wmma_bf16f32(v16bf a, v16bf b, v8f c) {
  return __builtin_amdgcn_wmma_f32_16x16x32_bf16(false, a, false, b,
                                                 (short)0, c, false, false);
}

#if USE_TDM
// Issue a TDM 2D tile load: global (row-major, row_stride elems of 2B) -> LDS.
// D# built per CDNA5 ISA 8.3/8.4. Per-wave instruction; EXEC ignored.
__device__ __forceinline__ void tdm_load_2d_bf16(uint32_t lds_off, uint64_t gaddr,
                                                 uint32_t tile_w, uint32_t tile_h,
                                                 uint32_t row_stride) {
  u32x4 g0;
  g0[0] = 1u;                                            // count=1 (valid, user)
  g0[1] = lds_off;                                       // lds_addr (bytes)
  g0[2] = (uint32_t)(gaddr & 0xffffffffu);               // global_addr[31:0]
  g0[3] = (uint32_t)((gaddr >> 32) & 0x01ffffffu)        // global_addr[56:32]
        | (2u << 30);                                    // type = 2 ("image")
  const uint32_t td0 = 1u << 20;                         // huge tensor dims:
  const uint32_t td1 = 1u << 20;                         // tiles always in-bounds
  i32x8 g1;
  g1[0] = (int)(1u << 16);                               // data_size=1 (2 bytes)
  g1[1] = (int)((td0 & 0xffffu) << 16);                  // tensor_dim0[15:0]
  g1[2] = (int)(((td0 >> 16) & 0xffffu) | ((td1 & 0xffffu) << 16));
  g1[3] = (int)(((td1 >> 16) & 0xffffu) | (tile_w << 16)); // tile_dim0
  g1[4] = (int)(tile_h & 0xffffu);                       // tile_dim1 (dim2=0)
  g1[5] = (int)row_stride;                               // tensor_dim0_stride
  g1[6] = 0;
  g1[7] = 0;
  i32x4 gz = {0, 0, 0, 0};
#if __clang_major__ >= 23
  i32x8 gz8 = {0, 0, 0, 0, 0, 0, 0, 0};
  __builtin_amdgcn_tensor_load_to_lds(g0, g1, gz, gz, gz8, 0);
#else
  __builtin_amdgcn_tensor_load_to_lds(g0, g1, gz, gz, 0);
#endif
}
#endif

// ---------------------------------------------------------------- converts
__global__ void k_f32_to_bf16(const float* __restrict__ src,
                              __bf16* __restrict__ dst, int n4) {
  int i = blockIdx.x * blockDim.x + threadIdx.x;
  if (i >= n4) return;
  float4 v = ((const float4*)src)[i];
  __align__(8) __bf16 o[4] = { (__bf16)v.x, (__bf16)v.y, (__bf16)v.z, (__bf16)v.w };
  ((uint2*)dst)[i] = *(const uint2*)o;
}

__global__ void k_transpose_f32_to_bf16(const float* __restrict__ src,
                                        __bf16* __restrict__ dst, int R, int C) {
  __shared__ float tile[32][33];
  int c0 = blockIdx.x * 32, r0 = blockIdx.y * 32;
  int tx = threadIdx.x, ty = threadIdx.y;        // block (32,8)
#pragma unroll
  for (int s = 0; s < 32; s += 8)
    tile[ty + s][tx] = src[(size_t)(r0 + ty + s) * C + (c0 + tx)];
  __syncthreads();
#pragma unroll
  for (int s = 0; s < 32; s += 8)
    dst[(size_t)(c0 + ty + s) * R + (r0 + tx)] = (__bf16)tile[tx][ty + s];
}

// ------------------------------------------------------------------- GEMM
// C[M,N] = A[M,K] * B[K,N] (+bias), with B supplied transposed: BT[N,K].
// 128x128 tile / workgroup, 8 waves as 4x2, each wave 32x64 (2x4 WMMA tiles).
template <bool F32OUT>
__global__ __launch_bounds__(256) void k_gemm_bf16(
    const __bf16* __restrict__ A, const __bf16* __restrict__ BT,
    const float* __restrict__ bias, void* __restrict__ Cout,
    int M, int N, int K) {
  const int lane = threadIdx.x & 31;
  const int wave = threadIdx.x >> 5;
  const int wm = wave >> 1, wn = wave & 1;
  const int m0 = blockIdx.y * 128 + wm * 32;
  const int n0 = blockIdx.x * 128 + wn * 64;
  v8f acc[2][4] = {};
  for (int k0 = 0; k0 < K; k0 += 32) {
    v16bf a0 = load_frag(A, m0,      K, k0, lane);
    v16bf a1 = load_frag(A, m0 + 16, K, k0, lane);
    if (k0 + 128 < K)
      __builtin_prefetch(A + (size_t)(m0 + (lane & 15)) * K + k0 + 128, 0, 1);
#pragma unroll
    for (int in = 0; in < 4; ++in) {
      v16bf b = load_frag(BT, n0 + in * 16, K, k0, lane);
      acc[0][in] = wmma_bf16f32(a0, b, acc[0][in]);
      acc[1][in] = wmma_bf16f32(a1, b, acc[1][in]);
    }
  }
  const int hi = (lane >> 4) & 1, cl = lane & 15;
#pragma unroll
  for (int im = 0; im < 2; ++im)
#pragma unroll
    for (int in = 0; in < 4; ++in)
#pragma unroll
      for (int r = 0; r < 8; ++r) {
        int row = m0 + im * 16 + r + 8 * hi;
        int col = n0 + in * 16 + cl;
        float v = acc[im][in][r] + bias[col];
        if (F32OUT) ((float*)Cout)[(size_t)row * N + col] = v;
        else        ((__bf16*)Cout)[(size_t)row * N + col] = (__bf16)v;
      }
}

// --------------------------------------------------------------- rotary
// In-place rotary on [b,n,h*d] bf16; each thread owns a (d, d+64) pair.
__global__ void k_rotary(__bf16* __restrict__ t, float scale) {
  int g = blockIdx.x * blockDim.x + threadIdx.x;
  int d = g & 63;
  int h = (g >> 6) & (HEADS - 1);
  int i = (g >> 10) & (SEQ - 1);
  int b = g >> 21;
  size_t base = (size_t)(b * SEQ + i) * DIM + h * DHEAD;
  float t1 = (float)t[base + d];
  float t2 = (float)t[base + d + 64];
  // inv_freq = 10000^(-d/64) = exp(-d * ln(10000)/64)
  float freq = (float)i * __expf(-(float)d * (9.210340371976184f / 64.0f));
  float sn, cs;
  __sincosf(freq, &sn, &cs);
  t[base + d]      = (__bf16)((t1 * cs - t2 * sn) * scale);
  t[base + d + 64] = (__bf16)((t2 * cs + t1 * sn) * scale);
}

// ---------------------------------------------------------- flash attention
// grid (n/128, heads, batch); 8 waves, wave w owns query rows i0+16w..+15.
// K block staged by the Tensor Data Mover (TENSORcnt), V transposed manually,
// online softmax in f32, QK^T and P@V on the bf16 WMMA pipe.
__global__ __launch_bounds__(256) void k_flash_attn(
    const __bf16* __restrict__ q, const __bf16* __restrict__ k,
    const __bf16* __restrict__ v, __bf16* __restrict__ ao) {
  __shared__ __align__(16) __bf16 sK[64 * DHEAD];    // [j][d]   16 KB
  __shared__ __align__(16) __bf16 sVt[DHEAD * 64];   // [d][j]   16 KB
  __shared__ __align__(16) __bf16 sP[8][16 * 64];    // per-wave 16 KB
  const int lane = threadIdx.x & 31;
  const int wave = threadIdx.x >> 5;
  const int it = blockIdx.x, h = blockIdx.y, b = blockIdx.z;
  const int i0 = it * 128;
  const int hi = lane >> 4, cl = lane & 15;

  const __bf16* qh = q + (size_t)b * SEQ * DIM + h * DHEAD;
  const __bf16* kh = k + (size_t)b * SEQ * DIM + h * DHEAD;
  const __bf16* vh = v + (size_t)b * SEQ * DIM + h * DHEAD;

  v16bf qf[4];
#pragma unroll
  for (int kc = 0; kc < 4; ++kc)
    qf[kc] = load_frag(qh, i0 + wave * 16, DIM, kc * 32, lane);

  v8f o[8] = {};
  float ms[8], ls[8];
#pragma unroll
  for (int r = 0; r < 8; ++r) { ms[r] = -1e30f; ls[r] = 0.f; }

  const int njb = 2 * it + 2;                 // causal: keys up to i0+127
  for (int jb = 0; jb < njb; ++jb) {
    const int j0 = jb * 64;
    __syncthreads();
#if USE_TDM
    if (wave == 0) {                          // one TDM issue per workgroup
      tdm_load_2d_bf16((uint32_t)(uintptr_t)(void*)sK,
                       (uint64_t)(uintptr_t)(kh + (size_t)j0 * DIM),
                       DHEAD, 64, DIM);
    }
#else
    for (int t = threadIdx.x; t < 64 * 16; t += 256) {      // K block, b128
      int r = t >> 4, s = t & 15;
      *(uint4*)(sK + r * DHEAD + s * 8) =
          *(const uint4*)(kh + (size_t)(j0 + r) * DIM + s * 8);
    }
#endif
    for (int t = threadIdx.x; t < 64 * DHEAD; t += 256) {   // V transposed
      int j = t >> 7, d = t & (DHEAD - 1);
      sVt[d * 64 + j] = vh[(size_t)(j0 + j) * DIM + d];
    }
#if USE_TDM
    if (wave == 0) __builtin_amdgcn_s_wait_tensorcnt(0);
#endif
    __syncthreads();

    // S = q @ k^T  (4 tiles of 16 cols)
    v8f st[4];
#pragma unroll
    for (int jt = 0; jt < 4; ++jt) {
      v8f sa = {};
#pragma unroll
      for (int kc = 0; kc < 4; ++kc) {
        v16bf bf = load_frag(sK, jt * 16, DHEAD, kc * 32, lane);
        sa = wmma_bf16f32(qf[kc], bf, sa);
      }
      st[jt] = sa;
    }
    if (j0 + 64 > i0) {                       // diagonal blocks: causal mask
#pragma unroll
      for (int jt = 0; jt < 4; ++jt) {
        int jg = j0 + jt * 16 + cl;
#pragma unroll
        for (int r = 0; r < 8; ++r) {
          int ig = i0 + wave * 16 + r + 8 * hi;
          if (jg > ig) st[jt][r] = -1e30f;
        }
      }
    }
    // online softmax: row max / rescale (rows live on 16-lane halves)
    float alpha[8];
#pragma unroll
    for (int r = 0; r < 8; ++r) {
      float mv = fmaxf(fmaxf(st[0][r], st[1][r]), fmaxf(st[2][r], st[3][r]));
      mv = fmaxf(mv, __shfl_xor(mv, 1));
      mv = fmaxf(mv, __shfl_xor(mv, 2));
      mv = fmaxf(mv, __shfl_xor(mv, 4));
      mv = fmaxf(mv, __shfl_xor(mv, 8));
      float mn = fmaxf(ms[r], mv);
      alpha[r] = __expf(ms[r] - mn);
      ms[r] = mn;
    }
    __bf16* sPw = &sP[wave][0];
#pragma unroll
    for (int jt = 0; jt < 4; ++jt)
#pragma unroll
      for (int r = 0; r < 8; ++r) {
        float p = __expf(st[jt][r] - ms[r]);
        st[jt][r] = p;
        sPw[(r + 8 * hi) * 64 + jt * 16 + cl] = (__bf16)p;   // re-layout via LDS
      }
#pragma unroll
    for (int r = 0; r < 8; ++r) {
      float rs = st[0][r] + st[1][r] + st[2][r] + st[3][r];
      rs += __shfl_xor(rs, 1);
      rs += __shfl_xor(rs, 2);
      rs += __shfl_xor(rs, 4);
      rs += __shfl_xor(rs, 8);
      ls[r] = ls[r] * alpha[r] + rs;
    }
#pragma unroll
    for (int dt = 0; dt < 8; ++dt)
#pragma unroll
      for (int r = 0; r < 8; ++r) o[dt][r] *= alpha[r];

    // O += P @ V
    v16bf pf0 = load_frag(sPw, 0, 64, 0, lane);
    v16bf pf1 = load_frag(sPw, 0, 64, 32, lane);
#pragma unroll
    for (int dt = 0; dt < 8; ++dt) {
      v16bf vf0 = load_frag(sVt, dt * 16, 64, 0, lane);
      o[dt] = wmma_bf16f32(pf0, vf0, o[dt]);
      v16bf vf1 = load_frag(sVt, dt * 16, 64, 32, lane);
      o[dt] = wmma_bf16f32(pf1, vf1, o[dt]);
    }
  }
#pragma unroll
  for (int r = 0; r < 8; ++r) {
    float inv = 1.0f / ls[r];
    size_t row = (size_t)b * SEQ + i0 + wave * 16 + r + 8 * hi;
    __bf16* orow = ao + row * DIM + h * DHEAD;
#pragma unroll
    for (int dt = 0; dt < 8; ++dt)
      orow[dt * 16 + cl] = (__bf16)(o[dt][r] * inv);
  }
}

// ---------------------------------------------------------------- launcher
extern "C" void kernel_launch(void* const* d_in, const int* in_sizes, int n_in,
                              void* d_out, int out_size, void* d_ws, size_t ws_size,
                              hipStream_t stream) {
  const float* x  = (const float*)d_in[0];
  const float* Wq = (const float*)d_in[1];
  const float* bq = (const float*)d_in[2];
  const float* Wk = (const float*)d_in[3];
  const float* bk = (const float*)d_in[4];
  const float* Wv = (const float*)d_in[5];
  const float* bv = (const float*)d_in[6];
  const float* Wo = (const float*)d_in[7];
  const float* bo = (const float*)d_in[8];
  float* out = (float*)d_out;

  char* ws = (char*)d_ws;
  const size_t SZ_X = (size_t)MROWS * DIM * 2;   // 16 MB per activation
  const size_t SZ_W = (size_t)DIM * DIM * 2;     //  8 MB per weight
  __bf16* xb  = (__bf16*)(ws);
  __bf16* WqT = (__bf16*)(ws + SZ_X);
  __bf16* WkT = (__bf16*)(ws + SZ_X + 1 * SZ_W);
  __bf16* WvT = (__bf16*)(ws + SZ_X + 2 * SZ_W);
  __bf16* WoT = (__bf16*)(ws + SZ_X + 3 * SZ_W);
  __bf16* qb  = (__bf16*)(ws + 1 * SZ_X + 4 * SZ_W);
  __bf16* kb  = (__bf16*)(ws + 2 * SZ_X + 4 * SZ_W);
  __bf16* vb  = (__bf16*)(ws + 3 * SZ_X + 4 * SZ_W);
  __bf16* ao  = (__bf16*)(ws + 4 * SZ_X + 4 * SZ_W);  // total 112 MB

  // 1) x -> bf16
  k_f32_to_bf16<<<(MROWS * DIM / 4 + 255) / 256, 256, 0, stream>>>(x, xb, MROWS * DIM / 4);
  // 2) weights -> bf16 transposed
  dim3 tb(32, 8), tg(DIM / 32, DIM / 32);
  k_transpose_f32_to_bf16<<<tg, tb, 0, stream>>>(Wq, WqT, DIM, DIM);
  k_transpose_f32_to_bf16<<<tg, tb, 0, stream>>>(Wk, WkT, DIM, DIM);
  k_transpose_f32_to_bf16<<<tg, tb, 0, stream>>>(Wv, WvT, DIM, DIM);
  k_transpose_f32_to_bf16<<<tg, tb, 0, stream>>>(Wo, WoT, DIM, DIM);
  // 3) q/k/v projections (WMMA GEMM)
  dim3 gg(DIM / 128, MROWS / 128);
  k_gemm_bf16<false><<<gg, 256, 0, stream>>>(xb, WqT, bq, qb, MROWS, DIM, DIM);
  k_gemm_bf16<false><<<gg, 256, 0, stream>>>(xb, WkT, bk, kb, MROWS, DIM, DIM);
  k_gemm_bf16<false><<<gg, 256, 0, stream>>>(xb, WvT, bv, vb, MROWS, DIM, DIM);
  // 4) rotary (+ q scale = 128^-0.5)
  const int rot_blocks = (BATCH * SEQ * HEADS * 64) / 256;
  k_rotary<<<rot_blocks, 256, 0, stream>>>(qb, 0.08838834764831845f);
  k_rotary<<<rot_blocks, 256, 0, stream>>>(kb, 1.0f);
  // 5) causal flash attention (WMMA + TDM staging)
  k_flash_attn<<<dim3(SEQ / 128, HEADS, BATCH), 256, 0, stream>>>(qb, kb, vb, ao);
  // 6) output projection -> f32 + bias
  k_gemm_bf16<true><<<gg, 256, 0, stream>>>(ao, WoT, bo, out, MROWS, DIM, DIM);

  (void)in_sizes; (void)n_in; (void)out_size; (void)ws_size;
}
